// GR_Weight_69277822484499
// MI455X (gfx1250) — matrix operand order repaired and checked by reading, not compiled, not run on hardware
//
#include <hip/hip_runtime.h>

// ---------------------------------------------------------------------------
// T=512, B=256, D=256, fp32 in/out.
//   * bf16 WMMA (f32 accum) everywhere: v_wmma_f32_16x16x32_bf16
//   * A-fragments register-resident, reused across GEMMs sharing an operand
//   * 512-step recurrence = ONE kernel (batch rows independent), h in LDS+regs
//   * k_fj stages its per-timestep A tiles with the Tensor Data Mover,
//     double-buffered via TENSORcnt (falls back to ds staging if the
//     builtin is absent)
// ---------------------------------------------------------------------------

#define T_ 512
#define B_ 256
#define D_ 256
#define BD_ (B_ * D_)               // 65536
#define MAT_ 65536                  // D*D
#define TBD_ ((size_t)T_ * B_ * D_) // 33,554,432

typedef __attribute__((ext_vector_type(16))) __bf16 v16bf;
typedef __attribute__((ext_vector_type(8)))  float  v8f;

union Frag { v16bf v; unsigned int u[8]; };

__device__ __forceinline__ float sigmoidf_(float x) { return 1.0f / (1.0f + __expf(-x)); }

__device__ __forceinline__ v8f vzero8() {
  v8f z = {0.f, 0.f, 0.f, 0.f, 0.f, 0.f, 0.f, 0.f};
  return z;
}

__device__ __forceinline__ v8f wmma_bf16(v16bf a, v16bf b, v8f c) {
  return __builtin_amdgcn_wmma_f32_16x16x32_bf16(false, a, false, b, (short)0, c, false, false);
}

// ---- LDS A-tile: 16 rows x 256 bf16, padded row stride 129 dwords ---------
#define LDS_TILE_DW (16 * 129)

__device__ __forceinline__ void stage_a_tile_bf16(const __bf16* gA, unsigned int* lA) {
  const unsigned int* g = (const unsigned int*)gA;   // 128 dwords per row
  int lane = threadIdx.x & 31;
  for (int it = 0; it < 64; ++it) {
    int idx = (it << 5) + lane;
    int r = idx >> 7, c = idx & 127;
    lA[r * 129 + c] = g[r * 128 + c];
  }
}

__device__ __forceinline__ void stage_a_tile_f32(const float* gA, unsigned int* lA) {
  int lane = threadIdx.x & 31;
  for (int it = 0; it < 64; ++it) {
    int idx = (it << 5) + lane;
    int r = idx >> 7, c = idx & 127;                 // c = bf16 pair index
    float f0 = gA[r * 256 + 2 * c];
    float f1 = gA[r * 256 + 2 * c + 1];
    unsigned short b0 = __builtin_bit_cast(unsigned short, (__bf16)f0);
    unsigned short b1 = __builtin_bit_cast(unsigned short, (__bf16)f1);
    lA[r * 129 + c] = (unsigned int)b0 | ((unsigned int)b1 << 16);
  }
}

// A fragment (ISA 16-bit A 16x32 layout), read from padded LDS tile
__device__ __forceinline__ v16bf load_a_frag_lds(const unsigned int* lA, int kbase) {
  Frag f; int lane = threadIdx.x & 31;
  int half = lane >> 4, m = lane & 15;
#pragma unroll
  for (int v = 0; v < 8; ++v) {
    int k0 = ((v & 3) << 1) + ((v >> 2) << 4) + (half << 3);
    f.u[v] = lA[m * 129 + ((kbase + k0) >> 1)];
  }
  return f.v;
}

// all 8 K-chunks of the A operand, kept in registers for reuse
struct AFrags { v16bf f[8]; };
__device__ __forceinline__ void load_a_frags(const unsigned int* lA, AFrags& A) {
#pragma unroll
  for (int kk = 0; kk < 8; ++kk) A.f[kk] = load_a_frag_lds(lA, kk << 5);
}

// B fragment for C = A @ W^T, W row-major [N,K] (ldw=256): contiguous pairs
__device__ __forceinline__ v16bf load_b_frag(const __bf16* Wn0, int kbase) {
  Frag f; int lane = threadIdx.x & 31;
  int half = lane >> 4, n = lane & 15;
  const __bf16* base = Wn0 + (size_t)n * 256 + kbase + (half << 3);
#pragma unroll
  for (int v = 0; v < 8; ++v)
    f.u[v] = *(const unsigned int*)(base + ((v & 3) << 1) + ((v >> 2) << 4));
  return f.v;
}

// 16x16 tile over K=256: B preloaded in chunks of 4 -> load clause, one wait,
// 4 chained WMMAs (instead of wait-per-wmma)
__device__ __forceinline__ v8f gemm_tile_preA(const AFrags& A, const __bf16* Wn0, v8f acc) {
#pragma unroll
  for (int c = 0; c < 2; ++c) {
    v16bf b0 = load_b_frag(Wn0, (c * 4 + 0) << 5);
    v16bf b1 = load_b_frag(Wn0, (c * 4 + 1) << 5);
    v16bf b2 = load_b_frag(Wn0, (c * 4 + 2) << 5);
    v16bf b3 = load_b_frag(Wn0, (c * 4 + 3) << 5);
    acc = wmma_bf16(A.f[c * 4 + 0], b0, acc);
    acc = wmma_bf16(A.f[c * 4 + 1], b1, acc);
    acc = wmma_bf16(A.f[c * 4 + 2], b2, acc);
    acc = wmma_bf16(A.f[c * 4 + 3], b3, acc);
  }
  return acc;
}

// ---------------------------------------------------------------------------
// Tensor Data Mover staging: 16x256 bf16 tile -> LDS with hardware padding
// (pad 1 dword every 128 dwords == our 129-dword conflict-free stride)
// ---------------------------------------------------------------------------
#if __has_builtin(__builtin_amdgcn_tensor_load_to_lds) && \
    __has_builtin(__builtin_amdgcn_s_wait_tensorcnt)
#define USE_TDM 1
typedef __attribute__((ext_vector_type(4))) unsigned int v4u_;
typedef __attribute__((ext_vector_type(8))) int v8i_;
typedef __attribute__((ext_vector_type(4))) int v4i_;

__device__ __forceinline__ void tdm_load_tile16x256(const __bf16* gsrc, unsigned int* ldst) {
  unsigned long long ga = (unsigned long long)(const void*)gsrc;
  unsigned int ga_lo = (unsigned int)__builtin_amdgcn_readfirstlane((int)(unsigned int)ga);
  unsigned int ga_hi = (unsigned int)__builtin_amdgcn_readfirstlane((int)(unsigned int)(ga >> 32));
  unsigned int lds   = (unsigned int)__builtin_amdgcn_readfirstlane(
                         (int)(unsigned int)(unsigned long long)(const void*)ldst);
  // D# group0: count=1 | lds_addr | global_addr[56:0] | type=2
  v4u_ g0 = { 1u, lds, ga_lo, (ga_hi & 0x01FFFFFFu) | 0x80000000u };
  // D# group1: data_size=2B(1), pad_enable, pad_interval=6(->128dw), pad_amount=0(->1dw)
  //   tensor_dim0=256 @bit48, tensor_dim1=16 @bit80, tile_dim0=256 @bit112,
  //   tile_dim1=16 @bit128, tensor_dim0_stride=256 @bit160
  v8i_ g1 = { (int)0x01910000u,
              (int)(256u << 16),
              (int)(16u << 16),
              (int)(256u << 16),
              16,
              256,
              0, 0 };
  v4i_ gz = { 0, 0, 0, 0 };
#if defined(__clang_major__) && (__clang_major__ >= 23)
  v8i_ gz8 = { 0, 0, 0, 0, 0, 0, 0, 0 };
  __builtin_amdgcn_tensor_load_to_lds(g0, g1, gz, gz, gz8, 0);
#else
  __builtin_amdgcn_tensor_load_to_lds(g0, g1, gz, gz, 0);
#endif
}
#else
#define USE_TDM 0
#endif

// ---------------------------------------------------------------------------
// Prep kernels
// ---------------------------------------------------------------------------
struct W14 { const float* p[14]; };
__global__ void k_cast_weights(W14 w, __bf16* dst) {
  int i = blockIdx.x * blockDim.x + threadIdx.x;
  if (i >= 14 * MAT_) return;
  dst[i] = (__bf16)w.p[i >> 16][i & 65535];
}

struct V8S { const float* mat[8]; const float* bias[8]; };
__global__ void k_prep_vec(V8S s, float* vec) {
  int v = blockIdx.x, n = threadIdx.x;
  const float* row = s.mat[v] + n * 256;
  float sum = s.bias[v][n];
  for (int k = 0; k < 256; ++k) sum += row[k];
  vec[v * 256 + n] = sum;
}

// ---------------------------------------------------------------------------
// Big time-parallel GEMM: C[M,256] = A[M,256] @ W[256,256]^T (+bias[n])
// ---------------------------------------------------------------------------
__global__ void k_gemm_big(const void* A, int aF32, const __bf16* W, const float* bias,
                           float* Cf, __bf16* Cb, int mtiles) {
  extern __shared__ unsigned int smem[];
  int wave = blockIdx.x * (blockDim.x >> 5) + (threadIdx.x >> 5);
  if (wave >= (mtiles << 4)) return;
  unsigned int* lA = smem + (threadIdx.x >> 5) * LDS_TILE_DW;
  int i = wave >> 4, j = wave & 15;
  if (aF32) stage_a_tile_f32((const float*)A + (size_t)i * 4096, lA);
  else      stage_a_tile_bf16((const __bf16*)A + (size_t)i * 4096, lA);
  AFrags Af; load_a_frags(lA, Af);
  v8f acc = gemm_tile_preA(Af, W + (size_t)j * 4096, vzero8());
  int lane = threadIdx.x & 31, half = lane >> 4;
  int n = j * 16 + (lane & 15);
  float bv = bias ? bias[n] : 0.f;
#pragma unroll
  for (int v = 0; v < 8; ++v) {
    size_t m = (size_t)i * 16 + v + (half << 3);
    size_t idx = m * 256 + n;
    float val = acc[v] + bv;
    if (Cf) Cf[idx] = val;
    if (Cb) Cb[idx] = (__bf16)val;
  }
}

// ---------------------------------------------------------------------------
// Fused 512-step recurrence. Block b owns batch rows [16b,16b+16).
// h_prev A-fragments loaded ONCE per step, reused by all 8 gate GEMMs.
// ---------------------------------------------------------------------------
struct RecPtrs {
  const __bf16 *Whz, *Whi, *Whr, *Who, *Whh;
  const __bf16 *PXz, *PXi, *PXr, *PXh, *PXo;
  const float  *XPf;
  __bf16 *HSb, *CSb, *GMb;
};

__global__ void __launch_bounds__(256) k_recurrent(RecPtrs P) {
  extern __shared__ unsigned int smem[];
  unsigned int* hA = smem;                 // h_prev (16x256 bf16, padded)
  unsigned int* rh = smem + LDS_TILE_DW;   // r*h    (16x256 bf16, padded)
  __bf16* hAe = (__bf16*)hA;
  __bf16* rhe = (__bf16*)rh;

  const int wave = threadIdx.x >> 5;       // 0..7: column tiles 2w, 2w+1
  const int lane = threadIdx.x & 31;
  const int half = lane >> 4;
  const int j0 = wave * 2;
  const size_t rowBase = (size_t)blockIdx.x * 16;

  for (int idx = threadIdx.x; idx < LDS_TILE_DW; idx += blockDim.x) hA[idx] = 0u;

  float hp[2][8], hs[2][8];
#pragma unroll
  for (int q = 0; q < 2; ++q)
#pragma unroll
    for (int v = 0; v < 8; ++v) { hp[q][v] = 0.f; hs[q][v] = 0.f; }

  __syncthreads();

  for (int t = 0; t < T_; ++t) {
    size_t tOff = (size_t)t * BD_;
    if (t + 1 < T_) {
      size_t pf = (size_t)(t + 1) * BD_ + rowBase * 256;
      __builtin_prefetch(P.PXz + pf, 0, 1);
      __builtin_prefetch(P.PXi + pf, 0, 1);
      __builtin_prefetch(P.PXr + pf, 0, 1);
      __builtin_prefetch(P.PXh + pf, 0, 1);
      __builtin_prefetch(P.PXo + pf, 0, 1);
      __builtin_prefetch(P.XPf + pf, 0, 1);
    }

    AFrags Ah; load_a_frags(hA, Ah);       // one LDS read of h per step
    float zg[2][8], ig[2][8], og[2][8];

    // ---- stage 1: z,i,o,r gates; r*h -> LDS ----
#pragma unroll
    for (int q = 0; q < 2; ++q) {
      int j = j0 + q;
      int n = j * 16 + (lane & 15);
      {
        v8f a = gemm_tile_preA(Ah, P.Whz + (size_t)j * 4096, vzero8());
#pragma unroll
        for (int v = 0; v < 8; ++v) {
          size_t gidx = tOff + (rowBase + v + (half << 3)) * 256 + n;
          zg[q][v] = sigmoidf_(a[v] + (float)P.PXz[gidx]);
        }
      }
      {
        v8f a = gemm_tile_preA(Ah, P.Whi + (size_t)j * 4096, vzero8());
#pragma unroll
        for (int v = 0; v < 8; ++v) {
          size_t gidx = tOff + (rowBase + v + (half << 3)) * 256 + n;
          ig[q][v] = sigmoidf_(a[v] + (float)P.PXi[gidx]);
        }
      }
      {
        v8f a = gemm_tile_preA(Ah, P.Who + (size_t)j * 4096, vzero8());
#pragma unroll
        for (int v = 0; v < 8; ++v) {
          size_t gidx = tOff + (rowBase + v + (half << 3)) * 256 + n;
          og[q][v] = sigmoidf_(a[v] + (float)P.PXo[gidx]);
        }
      }
      {
        v8f a = gemm_tile_preA(Ah, P.Whr + (size_t)j * 4096, vzero8());
#pragma unroll
        for (int v = 0; v < 8; ++v) {
          int m = v + (half << 3);
          size_t gidx = tOff + (rowBase + m) * 256 + n;
          float r = sigmoidf_(a[v] + (float)P.PXr[gidx]);
          rhe[m * 258 + n] = (__bf16)(r * hp[q][v]);
        }
      }
    }
    __syncthreads();

    // ---- stage 2: h~, cell, output ----
    AFrags Ar; load_a_frags(rh, Ar);
#pragma unroll
    for (int q = 0; q < 2; ++q) {
      int j = j0 + q;
      int n = j * 16 + (lane & 15);
      v8f a = gemm_tile_preA(Ar, P.Whh + (size_t)j * 4096, vzero8());
#pragma unroll
      for (int v = 0; v < 8; ++v) {
        int m = v + (half << 3);
        size_t gidx = tOff + (rowBase + m) * 256 + n;
        float ht = tanhf(a[v] + (float)P.PXh[gidx]);
        float xp = P.XPf[gidx];
        float z = zg[q][v], ii = ig[q][v], oo = og[q][v];
        float c = (1.f - z) * xp + z * ht + ii * hp[q][v];
        float hn = oo * tanhf(c);
        P.CSb[gidx] = (__bf16)c;
        P.HSb[gidx] = (__bf16)hn;
        hp[q][v] = hn;
        hs[q][v] += hn;
        hAe[m * 258 + n] = (__bf16)hn;
      }
    }
    __syncthreads();
  }

#pragma unroll
  for (int q = 0; q < 2; ++q) {
    int n = (j0 + q) * 16 + (lane & 15);
#pragma unroll
    for (int v = 0; v < 8; ++v)
      P.GMb[(rowBase + v + (half << 3)) * 256 + n] = (__bf16)(hs[q][v] * (1.0f / T_));
  }
}

// ---------------------------------------------------------------------------
// f_j = sum_t sigmoid(hs_t @ g_hh^T + gbh) * cs_t
// A tiles streamed by the Tensor Data Mover, double-buffered on TENSORcnt.
// 2 waves / 64-thread block (2 LDS buffers per wave).
// ---------------------------------------------------------------------------
__global__ void k_fj(const __bf16* HSb, const __bf16* CSb, const __bf16* Ghh,
                     const float* gbh, float* FJ) {
  extern __shared__ unsigned int smem[];
  int wave = blockIdx.x * (blockDim.x >> 5) + (threadIdx.x >> 5);
  unsigned int* buf0 = smem + (threadIdx.x >> 5) * (2 * LDS_TILE_DW);
  unsigned int* buf1 = buf0 + LDS_TILE_DW;
  int i = wave >> 4, j = wave & 15;
  int lane = threadIdx.x & 31, half = lane >> 4;
  int n = j * 16 + (lane & 15);
  const __bf16* Wn0 = Ghh + (size_t)j * 4096;
  float bv = gbh[n];
  float fj[8] = {0.f, 0.f, 0.f, 0.f, 0.f, 0.f, 0.f, 0.f};

#if USE_TDM
  tdm_load_tile16x256(HSb + (size_t)i * 4096, buf0);
#endif
  for (int t = 0; t < T_; ++t) {
    unsigned int* cur = (t & 1) ? buf1 : buf0;
#if USE_TDM
    if (t + 1 < T_) {
      unsigned int* nxt = (t & 1) ? buf0 : buf1;
      tdm_load_tile16x256(HSb + (size_t)(t + 1) * BD_ + (size_t)i * 4096, nxt);
      __builtin_amdgcn_s_wait_tensorcnt(1);   // current tile landed
    } else {
      __builtin_amdgcn_s_wait_tensorcnt(0);
    }
#else
    stage_a_tile_bf16(HSb + (size_t)t * BD_ + (size_t)i * 4096, cur);
#endif
    AFrags Af; load_a_frags(cur, Af);
    v8f acc = gemm_tile_preA(Af, Wn0, vzero8());
    size_t tOff = (size_t)t * BD_;
#pragma unroll
    for (int v = 0; v < 8; ++v) {
      size_t m = (size_t)i * 16 + v + (half << 3);
      fj[v] += sigmoidf_(acc[v] + bv) * (float)CSb[tOff + m * 256 + n];
    }
  }
#pragma unroll
  for (int v = 0; v < 8; ++v)
    FJ[((size_t)i * 16 + v + (half << 3)) * 256 + n] = fj[v];
}

// ---------------------------------------------------------------------------
// f_g / o_g from g_mean (two GEMMs sharing register-resident A), final output
// ---------------------------------------------------------------------------
__global__ void k_final(const __bf16* GMb, const __bf16* Ghc, const __bf16* Gho,
                        const float* gbg, const float* gbo, const float* FJ, float* out) {
  extern __shared__ unsigned int smem[];
  int wave = blockIdx.x * (blockDim.x >> 5) + (threadIdx.x >> 5);
  unsigned int* lA = smem + (threadIdx.x >> 5) * LDS_TILE_DW;
  int i = wave >> 4, j = wave & 15;
  int lane = threadIdx.x & 31, half = lane >> 4;
  int n = j * 16 + (lane & 15);
  stage_a_tile_bf16(GMb + (size_t)i * 4096, lA);
  AFrags Af; load_a_frags(lA, Af);
  v8f aF = gemm_tile_preA(Af, Ghc + (size_t)j * 4096, vzero8());
  v8f aO = gemm_tile_preA(Af, Gho + (size_t)j * 4096, vzero8());
  float bg = gbg[n], bo = gbo[n];
#pragma unroll
  for (int v = 0; v < 8; ++v) {
    size_t idx = ((size_t)i * 16 + v + (half << 3)) * 256 + n;
    float fg = sigmoidf_(aF[v] + bg);
    float oo = sigmoidf_(aO[v] + bo);
    float cg = FJ[idx] + fg;
    out[idx] = oo * tanhf(cg);
    out[(size_t)BD_ + idx] = cg;
  }
}

// ---------------------------------------------------------------------------
// host
// ---------------------------------------------------------------------------
extern "C" void kernel_launch(void* const* d_in, const int* in_sizes, int n_in,
                              void* d_out, int out_size, void* d_ws, size_t ws_size,
                              hipStream_t stream) {
  (void)in_sizes; (void)n_in; (void)out_size; (void)ws_size;

  const float* x    = (const float*)d_in[0];
  const float* a    = (const float*)d_in[1];
  const float* w_xz = (const float*)d_in[2];
  const float* w_hz = (const float*)d_in[3];
  const float* w_gz = (const float*)d_in[4];
  const float* w_xi = (const float*)d_in[5];
  const float* w_hi = (const float*)d_in[6];
  const float* w_gi = (const float*)d_in[7];
  const float* w_xr = (const float*)d_in[8];
  const float* w_hr = (const float*)d_in[9];
  const float* w_gr = (const float*)d_in[10];
  const float* w_xh = (const float*)d_in[11];
  const float* w_hh = (const float*)d_in[12];
  const float* w_gh = (const float*)d_in[13];
  const float* w_xo = (const float*)d_in[14];
  const float* w_ho = (const float*)d_in[15];
  const float* w_go = (const float*)d_in[16];
  const float* b_z  = (const float*)d_in[17];
  /* d_in[18] = b_i: unused — reference reuses b_z for the i-gate */
  const float* b_r  = (const float*)d_in[19];
  const float* b_h  = (const float*)d_in[20];
  const float* b_o  = (const float*)d_in[21];
  const float* g_hh = (const float*)d_in[22];
  const float* g_gh = (const float*)d_in[23];
  const float* g_hc = (const float*)d_in[24];
  const float* g_gc = (const float*)d_in[25];
  const float* g_ho = (const float*)d_in[26];
  const float* g_go = (const float*)d_in[27];
  const float* gb_h = (const float*)d_in[28];
  const float* gb_g = (const float*)d_in[29];
  const float* gb_o = (const float*)d_in[30];

  char* ws = (char*)d_ws;
  size_t off = 0;
  auto alloc = [&](size_t bytes) {
    size_t o = off; off += (bytes + 255) & ~(size_t)255; return o;
  };
  __bf16* WB  = (__bf16*)(ws + alloc((size_t)14 * MAT_ * 2));
  float*  VEC = (float*) (ws + alloc(8 * 256 * 4));
  float*  XPf = (float*) (ws + alloc(TBD_ * 4));
  __bf16* XPb = (__bf16*)(ws + alloc(TBD_ * 2));
  __bf16* PXz = (__bf16*)(ws + alloc(TBD_ * 2));
  __bf16* PXi = (__bf16*)(ws + alloc(TBD_ * 2));
  __bf16* PXr = (__bf16*)(ws + alloc(TBD_ * 2));
  __bf16* PXh = (__bf16*)(ws + alloc(TBD_ * 2));
  __bf16* PXo = (__bf16*)(ws + alloc(TBD_ * 2));
  __bf16* HSb = (__bf16*)(ws + alloc(TBD_ * 2));
  __bf16* CSb = (__bf16*)(ws + alloc(TBD_ * 2));
  float*  FJ  = (float*) (ws + alloc((size_t)BD_ * 4));
  __bf16* GMb = (__bf16*)(ws + alloc((size_t)BD_ * 2));

  __bf16 *Ab  = WB,            *Wxz = WB + 1*MAT_, *Wxi = WB + 2*MAT_,
         *Wxr = WB + 3*MAT_,   *Wxh = WB + 4*MAT_, *Wxo = WB + 5*MAT_,
         *Whz = WB + 6*MAT_,   *Whi = WB + 7*MAT_, *Whr = WB + 8*MAT_,
         *Whh = WB + 9*MAT_,   *Who = WB + 10*MAT_,
         *Ghh = WB + 11*MAT_,  *Ghc = WB + 12*MAT_, *Gho = WB + 13*MAT_;

  W14 w14 = {{ a, w_xz, w_xi, w_xr, w_xh, w_xo,
               w_hz, w_hi, w_hr, w_hh, w_ho,
               g_hh, g_hc, g_ho }};
  k_cast_weights<<<(14 * MAT_ + 255) / 256, 256, 0, stream>>>(w14, WB);

  V8S vs;
  const float* vmats[8] = { w_gz, w_gi, w_gr, w_gh, w_go, g_gh, g_gc, g_go };
  const float* vbias[8] = { b_z,  b_z,  b_r,  b_h,  b_o,  gb_h, gb_g, gb_o };
  for (int k = 0; k < 8; ++k) { vs.mat[k] = vmats[k]; vs.bias[k] = vbias[k]; }
  k_prep_vec<<<8, 256, 0, stream>>>(vs, VEC);

  const int mtiles = (T_ * B_) / 16;               // 8192
  const int bigBlocks = (mtiles * 16) / 4;         // 4 waves / 128-thr block
  const unsigned lds4 = 4 * LDS_TILE_DW * 4;       // 33,024 B

  k_gemm_big<<<bigBlocks, 128, lds4, stream>>>(x, 1, Ab, nullptr, XPf, XPb, mtiles);
  k_gemm_big<<<bigBlocks, 128, lds4, stream>>>(XPb, 0, Wxz, VEC + 0*256, nullptr, PXz, mtiles);
  k_gemm_big<<<bigBlocks, 128, lds4, stream>>>(XPb, 0, Wxi, VEC + 1*256, nullptr, PXi, mtiles);
  k_gemm_big<<<bigBlocks, 128, lds4, stream>>>(XPb, 0, Wxr, VEC + 2*256, nullptr, PXr, mtiles);
  k_gemm_big<<<bigBlocks, 128, lds4, stream>>>(XPb, 0, Wxh, VEC + 3*256, nullptr, PXh, mtiles);
  k_gemm_big<<<bigBlocks, 128, lds4, stream>>>(XPb, 0, Wxo, VEC + 4*256, nullptr, PXo, mtiles);

  RecPtrs rp;
  rp.Whz = Whz; rp.Whi = Whi; rp.Whr = Whr; rp.Who = Who; rp.Whh = Whh;
  rp.PXz = PXz; rp.PXi = PXi; rp.PXr = PXr; rp.PXh = PXh; rp.PXo = PXo;
  rp.XPf = XPf; rp.HSb = HSb; rp.CSb = CSb; rp.GMb = GMb;
  const unsigned lds2 = 2 * LDS_TILE_DW * 4;       // 16,512 B
  k_recurrent<<<B_ / 16, 256, lds2, stream>>>(rp);

  // 2 waves / 64-thr block, 2 LDS buffers each (double-buffered TDM)
  k_fj<<<(16 * 16) / 2, 64, 2 * 2 * LDS_TILE_DW * 4, stream>>>(HSb, CSb, Ghh, VEC + 5*256, FJ);
  k_final<<<(16 * 16) / 4, 128, lds4, stream>>>(GMb, Ghc, Gho, VEC + 6*256, VEC + 7*256,
                                                FJ, (float*)d_out);
}
